// QuerySAT_71545565217135
// MI455X (gfx1250) — matrix operand order repaired and checked by reading, not compiled, not run on hardware
//
#include <hip/hip_runtime.h>
#include <hip/hip_fp16.h>

// ---------------- types / constants ----------------
typedef __attribute__((ext_vector_type(16))) _Float16 v16h;
typedef __attribute__((ext_vector_type(8)))  float    v8f;

#define NV 20000
#define NC 84000
#define NG 32
#define NE (3*NC)
#define VPG (NV/NG)   // 625
#define CPG (NC/NG)   // 2625
#define RNDS 16

// gfx1250 async global->LDS path (ASYNCcnt), with safe fallback
#if defined(__has_builtin)
#  if __has_builtin(__builtin_amdgcn_global_load_async_to_lds_b128) && \
      __has_builtin(__builtin_amdgcn_s_wait_asynccnt)
#    define USE_ASYNC_LDS 1
#  endif
#endif
#ifndef USE_ASYNC_LDS
#  define USE_ASYNC_LDS 0
#endif

#if USE_ASYNC_LDS
// builtin signature (from hipcc diagnostic): (int4 AS1*, int4 AS3*, imm int, imm int)
typedef int v4i32 __attribute__((__vector_size__(4 * sizeof(int))));
typedef v4i32 __attribute__((address_space(1)))* gp1_t;
typedef v4i32 __attribute__((address_space(3)))* lp3_t;
#endif

__device__ __forceinline__ float softplusf(float x) {
  if (x > 20.f) return x;
  if (x < -20.f) return expf(x);
  return log1pf(expf(x));
}
__device__ __forceinline__ float lrelu_f(float x) { return x > 0.f ? x : 0.2f * x; }

// ---------------- WMMA GEMM ----------------
// C[M,N] = act( A_f16[M,K] @ W_f16[K,N] + bias )   (K = Kpad mult of 32; ldb = Npad mult of 16)
__global__ __launch_bounds__(256) void gemm_f16w(
    const _Float16* __restrict__ A, int lda,
    const _Float16* __restrict__ W, int ldb,
    const float* __restrict__ bias,
    float* __restrict__ C32, int ldc32,
    _Float16* __restrict__ C16, int ldc16,
    int M, int N, int K, int act)
{
  __shared__ _Float16 As[64 * 32] __attribute__((aligned(64)));
  __shared__ _Float16 Bs[32 * 128] __attribute__((aligned(64)));
  const int tid  = threadIdx.x;
  const int lane = tid & 31;
  const int wave = tid >> 5;
  const int wm = wave & 3;   // 0..3  (M sub-tile)
  const int wn = wave >> 2;  // 0..1  (N half)
  const long bm = (long)blockIdx.x * 64;
  const long bn = (long)blockIdx.y * 128;

  v8f acc0 = {}, acc1 = {}, acc2 = {}, acc3 = {};

  const int ar = tid >> 2;          // 0..63
  const int ac = (tid & 3) << 3;    // 0,8,16,24
  const int br = tid >> 3;          // 0..31
  const int bc = (tid & 7) << 4;    // 0..112

  const bool aIn = (bm + ar) < M;          // k-invariant
  const bool bIn = (bn + bc + 16) <= ldb;  // k-invariant (pad cols of W are zero)

#if USE_ASYNC_LDS
  { // pre-zero LDS lanes that never receive async data (stay zero for all k)
    float4 z = make_float4(0.f, 0.f, 0.f, 0.f);
    if (!aIn) *(float4*)(&As[ar * 32 + ac]) = z;
    if (!bIn) {
      *(float4*)(&Bs[br * 128 + bc])     = z;
      *(float4*)(&Bs[br * 128 + bc + 8]) = z;
    }
  }
#endif

  for (int k0 = 0; k0 < K; k0 += 32) {
#if USE_ASYNC_LDS
    // async global->LDS staging (tracked with ASYNCcnt, no VGPR bounce)
    if (aIn) {
      const _Float16* src = A + (bm + ar) * (long)lda + k0 + ac;
      __builtin_amdgcn_global_load_async_to_lds_b128(
          (gp1_t)src, (lp3_t)&As[ar * 32 + ac], 0, 0);
    }
    if (bIn) {
      const _Float16* src = W + (long)(k0 + br) * ldb + (bn + bc);
      __builtin_amdgcn_global_load_async_to_lds_b128(
          (gp1_t)src, (lp3_t)&Bs[br * 128 + bc], 0, 0);
      __builtin_amdgcn_global_load_async_to_lds_b128(
          (gp1_t)(src + 8), (lp3_t)&Bs[br * 128 + bc + 8], 0, 0);
    }
    __builtin_amdgcn_s_wait_asynccnt(0);
    __syncthreads();
#else
    { // A tile 64x32
      float4 v = make_float4(0.f, 0.f, 0.f, 0.f);
      if (aIn) {
        const _Float16* src = A + (bm + ar) * (long)lda + k0 + ac;
        v = *(const float4*)src;
        if (k0 + 32 < K) __builtin_prefetch((const void*)(src + 32), 0, 1);
      }
      *(float4*)(&As[ar * 32 + ac]) = v;
    }
    { // B tile 32x128
      const _Float16* src = W + (long)(k0 + br) * ldb + (bn + bc);
      if (bIn) {
        *(float4*)(&Bs[br * 128 + bc])     = *(const float4*)(src);
        *(float4*)(&Bs[br * 128 + bc + 8]) = *(const float4*)(src + 8);
        if (k0 + 32 < K) __builtin_prefetch((const void*)(src + 32L * ldb), 0, 1);
      } else {
        float4 z = make_float4(0.f, 0.f, 0.f, 0.f);
        *(float4*)(&Bs[br * 128 + bc])     = z;
        *(float4*)(&Bs[br * 128 + bc + 8]) = z;
      }
    }
    __syncthreads();
#endif
    // A frag: lanes 0-15 rows M, K 0..15; lanes 16-31 same rows, K 16..31
    const v16h a  = *(const v16h*)(&As[(wm * 16 + (lane & 15)) * 32 + ((lane >> 4) << 4)]);
    // B frag: lane = N column (lane&15), K 16*(lane>>4)+s
    v16h b0, b1, b2, b3;
    #pragma unroll
    for (int s = 0; s < 16; ++s) {
      const int bo = (((lane >> 4) << 4) + s) * 128 + wn * 64 + (lane & 15);
      b0[s] = Bs[bo +  0];
      b1[s] = Bs[bo + 16];
      b2[s] = Bs[bo + 32];
      b3[s] = Bs[bo + 48];
    }
    acc0 = __builtin_amdgcn_wmma_f32_16x16x32_f16(false, a, false, b0, (short)0, acc0, false, false);
    acc1 = __builtin_amdgcn_wmma_f32_16x16x32_f16(false, a, false, b1, (short)0, acc1, false, false);
    acc2 = __builtin_amdgcn_wmma_f32_16x16x32_f16(false, a, false, b2, (short)0, acc2, false, false);
    acc3 = __builtin_amdgcn_wmma_f32_16x16x32_f16(false, a, false, b3, (short)0, acc3, false, false);
    __syncthreads();
  }

  v8f accs[4] = {acc0, acc1, acc2, acc3};
  const long rbase = bm + wm * 16 + ((lane >> 4) << 3);  // lanes>=16 -> rows +8
  const int  cbl   = wn * 64 + (lane & 15);
  #pragma unroll
  for (int nt = 0; nt < 4; ++nt) {
    long col = bn + cbl + nt * 16;
    if (col < N) {
      float bv = bias ? bias[col] : 0.f;
      #pragma unroll
      for (int r = 0; r < 8; ++r) {
        long row = rbase + r;
        if (row < M) {
          float v = accs[nt][r] + bv;
          if (act) v = lrelu_f(v);
          if (C32) C32[row * (long)ldc32 + col] = v;
          if (C16) C16[row * (long)ldc16 + col] = (_Float16)v;
        }
      }
    }
  }
}

// ---------------- utility / elementwise kernels ----------------
__global__ void fill_f32_kernel(float* p, float v, long n) {
  long i = (long)blockIdx.x * blockDim.x + threadIdx.x;
  if (i < n) p[i] = v;
}
__global__ void fill_f16_kernel(_Float16* p, long n) {
  long i = (long)blockIdx.x * blockDim.x + threadIdx.x;
  if (i < n) p[i] = (_Float16)0.f;
}
__global__ void convert_w_kernel(const float* __restrict__ src, _Float16* __restrict__ dst,
                                 int K, int N, int Kpad, int Npad) {
  long i = (long)blockIdx.x * blockDim.x + threadIdx.x;
  long tot = (long)Kpad * Npad;
  if (i >= tot) return;
  int k = (int)(i / Npad), n = (int)(i % Npad);
  float v = (k < K && n < N) ? src[(long)k * N + n] : 0.f;
  dst[i] = (_Float16)v;
}
__global__ void cvt_h_kernel(const float* __restrict__ s, _Float16* __restrict__ d, long n) {
  long i = (long)blockIdx.x * blockDim.x + threadIdx.x;
  if (i < n) d[i] = (_Float16)s[i];
}

// degrees / weights (once per launch)
__global__ void deg_scatter_kernel(const int* __restrict__ lit, float* __restrict__ deg) {
  int e = blockIdx.x * blockDim.x + threadIdx.x;
  if (e < NE) atomicAdd(&deg[lit[e]], 1.f);
}
__global__ void deg_weights_kernel(const float* __restrict__ deg,
                                   float* __restrict__ dw, float* __restrict__ vdw) {
  int i = blockIdx.x * blockDim.x + threadIdx.x;
  if (i < 2 * NV) dw[i] = rsqrtf(fmaxf(deg[i], 1.f));
  if (i < NV)     vdw[i] = 4.f * rsqrtf(fmaxf(deg[i] + deg[i + NV], 1.f));
}

// v1 = [vars, noise] (f16, lda 160); varf16; unit cols 128..255 = vars
__global__ void build_v1_kernel(const float* __restrict__ vars, const float* __restrict__ noise_t,
                                _Float16* __restrict__ v1, _Float16* __restrict__ varh,
                                _Float16* __restrict__ unit) {
  int i = blockIdx.x * blockDim.x + threadIdx.x;
  if (i >= NV * 128) return;
  int v = i >> 7, j = i & 127;
  float x = vars[i];
  _Float16 h = (_Float16)x;
  v1[(long)v * 160 + j] = h;
  varh[i] = h;
  unit[(long)v * 512 + 128 + j] = h;
  if (j < 4) v1[(long)v * 160 + 128 + j] = (_Float16)noise_t[v * 4 + j];
}

// cl[c,j] = exp(-sum softplus(+/-q)); clause_unit cols 256..383 = 4*cl
__global__ void clause_val_kernel(const int* __restrict__ lit, const float* __restrict__ q,
                                  float* __restrict__ cl, _Float16* __restrict__ cu) {
  int i = blockIdx.x * blockDim.x + threadIdx.x;
  if (i >= NC * 128) return;
  int c = i >> 7, j = i & 127;
  float s = 0.f;
  #pragma unroll
  for (int e = 0; e < 3; ++e) {
    int l = lit[3 * c + e];
    float x = (l < NV) ? q[(long)l * 128 + j] : -q[(long)(l - NV) * 128 + j];
    s += softplusf(x);
  }
  float cv = expf(-s);
  cl[i] = cv;
  cu[(long)c * 384 + 256 + j] = (_Float16)(4.f * cv);
}

__global__ void scatter_cl_kernel(const int* __restrict__ lit, const float* __restrict__ cl,
                                  float* __restrict__ S) {
  long i = (long)blockIdx.x * blockDim.x + threadIdx.x;
  if (i >= (long)NE * 128) return;
  int e = (int)(i >> 7), j = (int)(i & 127);
  int c = e / 3;
  atomicAdd(&S[(long)lit[e] * 128 + j], cl[(long)c * 128 + j]);
}

// unit cols 0..127 = grad_q * var_degree_weight
__global__ void grad_kernel(const float* __restrict__ q, const float* __restrict__ S,
                            const float* __restrict__ vdw, _Float16* __restrict__ unit) {
  int i = blockIdx.x * blockDim.x + threadIdx.x;
  if (i >= NV * 128) return;
  int v = i >> 7, j = i & 127;
  float qq = q[i];
  float sp = 1.f / (1.f + expf(-qq));
  float sn = 1.f / (1.f + expf(qq));
  float g = (sn * S[(long)(v + NV) * 128 + j] - sp * S[i]) * vdw[v];
  unit[(long)v * 512 + j] = (_Float16)g;
}

// clause_unit cols 0..127 = state; 128..255 = gathered messages / sqrt(3)
__global__ void build_cu_kernel(const int* __restrict__ lit, const float* __restrict__ cstate,
                                const float* __restrict__ vmsg, _Float16* __restrict__ cu) {
  int i = blockIdx.x * blockDim.x + threadIdx.x;
  if (i >= NC * 128) return;
  int c = i >> 7, j = i & 127;
  cu[(long)c * 384 + j] = (_Float16)cstate[i];
  float m = 0.f;
  #pragma unroll
  for (int e = 0; e < 3; ++e) {
    int l = lit[3 * c + e];
    m += (l < NV) ? vmsg[(long)l * 256 + j] : vmsg[(long)(l - NV) * 256 + 128 + j];
  }
  cu[(long)c * 384 + 128 + j] = (_Float16)(m * 0.57735026919f); // * 1/sqrt(3)
}

// pair_norm over one contiguous graph per block; out = norm(x) + 0.1*old
__global__ __launch_bounds__(128) void pair_norm_kernel(const float* __restrict__ x, int ldx,
                                                        const float* __restrict__ old,
                                                        float* __restrict__ out, int rpg) {
  __shared__ float red[128];
  int g = blockIdx.x, j = threadIdx.x;
  const float* xg = x + (long)g * rpg * ldx;
  float s = 0.f;
  for (int r = 0; r < rpg; ++r) s += xg[(long)r * ldx + j];
  float mean = s / (float)rpg;
  float sq = 0.f;
  for (int r = 0; r < rpg; ++r) { float d = xg[(long)r * ldx + j] - mean; sq += d * d; }
  red[j] = sq; __syncthreads();
  for (int t = 64; t > 0; t >>= 1) { if (j < t) red[j] += red[j + t]; __syncthreads(); }
  float rs = rsqrtf(red[0] / ((float)rpg * 128.f) + 1e-6f);
  long base = (long)g * rpg * 128;
  for (int r = 0; r < rpg; ++r) {
    long o = base + (long)r * 128 + j;
    out[o] = (xg[(long)r * ldx + j] - mean) * rs + 0.1f * old[o];
  }
}

__global__ void scatter_vloss_kernel(const int* __restrict__ lit, const float* __restrict__ cdata,
                                     float* __restrict__ S2) {
  long i = (long)blockIdx.x * blockDim.x + threadIdx.x;
  if (i >= (long)NE * 128) return;
  int e = (int)(i >> 7), j = (int)(i & 127);
  int c = e / 3;
  atomicAdd(&S2[(long)lit[e] * 128 + j], cdata[(long)c * 256 + j]);
}

// unit cols 256..511 = vloss (pos/neg literal halves) * degree_weight
__global__ void vloss_kernel(const float* __restrict__ S2, const float* __restrict__ dw,
                             _Float16* __restrict__ unit) {
  int i = blockIdx.x * blockDim.x + threadIdx.x;
  if (i >= NV * 128) return;
  int v = i >> 7, j = i & 127;
  unit[(long)v * 512 + 256 + j] = (_Float16)(S2[i] * dw[v]);
  unit[(long)v * 512 + 384 + j] = (_Float16)(S2[(long)(v + NV) * 128 + j] * dw[v + NV]);
}

// per-clause logit loss -> per_graph[32][8]
__global__ void loss_clause_kernel(const int* __restrict__ lit, const float* __restrict__ logits,
                                   float* __restrict__ pg) {
  int c = blockIdx.x * blockDim.x + threadIdx.x;
  if (c >= NC) return;
  int g = c / CPG;
  int l0 = lit[3 * c], l1 = lit[3 * c + 1], l2 = lit[3 * c + 2];
  #pragma unroll
  for (int j = 0; j < 8; ++j) {
    float s = 0.f;
    s += softplusf(l0 < NV ? logits[(long)l0 * 8 + j] : -logits[(long)(l0 - NV) * 8 + j]);
    s += softplusf(l1 < NV ? logits[(long)l1 * 8 + j] : -logits[(long)(l1 - NV) * 8 + j]);
    s += softplusf(l2 < NV ? logits[(long)l2 * 8 + j] : -logits[(long)(l2 - NV) * 8 + j]);
    float cv = expf(-s);
    float pc = cv * (-logf(1.f - cv + 1e-10f));
    atomicAdd(&pg[g * 8 + j], pc);
  }
}

__global__ __launch_bounds__(32) void loss_fin_kernel(const float* __restrict__ pg,
                                                      float* __restrict__ out_loss) {
  __shared__ float tot[32];
  int g = threadIdx.x;
  float v[8];
  const float s0 = sqrtf(1e-6f);
  #pragma unroll
  for (int j = 0; j < 8; ++j) v[j] = sqrtf(pg[g * 8 + j] + 1e-6f) - s0;
  #pragma unroll
  for (int a = 0; a < 8; ++a)
    #pragma unroll
    for (int b = a + 1; b < 8; ++b)
      if (v[b] > v[a]) { float t = v[a]; v[a] = v[b]; v[b] = t; }  // descending
  float d = 0.f;
  #pragma unroll
  for (int k = 0; k < 8; ++k) d += v[k] * (float)((k + 1) * (k + 1));
  tot[g] = d; __syncthreads();
  for (int t = 16; t > 0; t >>= 1) { if (g < t) tot[g] += tot[g + t]; __syncthreads(); }
  if (g == 0) atomicAdd(out_loss, tot[0] / (204.f * (float)RNDS));
}

// ---------------- host ----------------
extern "C" void kernel_launch(void* const* d_in, const int* in_sizes, int n_in,
                              void* d_out, int out_size, void* d_ws, size_t ws_size,
                              hipStream_t stream) {
  (void)in_sizes; (void)n_in; (void)out_size; (void)ws_size;
  const int*   lit_idx = (const int*)d_in[0];
  const float* noise   = (const float*)d_in[4];
  // params: JAX pytree order (alphabetical keys; per layer 'b' then 'w')
  const float* cb0 = (const float*)d_in[5];   const float* cw0 = (const float*)d_in[6];
  const float* cb1 = (const float*)d_in[7];   const float* cw1 = (const float*)d_in[8];
  const float* lb0 = (const float*)d_in[9];   const float* lw0 = (const float*)d_in[10];
  const float* lb1 = (const float*)d_in[11];  const float* lw1 = (const float*)d_in[12];
  const float* lb2 = (const float*)d_in[13];  const float* lw2 = (const float*)d_in[14];
  const float* ub0 = (const float*)d_in[15];  const float* uw0 = (const float*)d_in[16];
  const float* ub1 = (const float*)d_in[17];  const float* uw1 = (const float*)d_in[18];
  const float* ub2 = (const float*)d_in[19];  const float* uw2 = (const float*)d_in[20];
  const float* ob0 = (const float*)d_in[21];  const float* ow0 = (const float*)d_in[22];
  const float* ob1 = (const float*)d_in[23];  const float* ow1 = (const float*)d_in[24];
  const float* qb0 = (const float*)d_in[25];  const float* qw0 = (const float*)d_in[26];
  const float* qb1 = (const float*)d_in[27];  const float* qw1 = (const float*)d_in[28];

  float* out = (float*)d_out;  // [20000*8 logits][1 mean loss]
  char* ws = (char*)d_ws;
  size_t off = 0;
  auto alloc = [&](size_t bytes) -> char* {
    char* p = ws + off;
    off = (off + bytes + 255) & ~(size_t)255;
    return p;
  };
  // f32 buffers
  float* variables   = (float*)alloc((size_t)NV * 128 * 4);
  float* newvars     = (float*)alloc((size_t)NV * 128 * 4);
  float* clause_st   = (float*)alloc((size_t)NC * 128 * 4);
  float* query       = (float*)alloc((size_t)NV * 128 * 4);
  float* clv         = (float*)alloc((size_t)NC * 128 * 4);
  float* var_msg     = (float*)alloc((size_t)NV * 256 * 4);
  float* clause_data = (float*)alloc((size_t)NC * 256 * 4);
  float* S           = (float*)alloc((size_t)2 * NV * 128 * 4);  // round-zeroed region start
  float* S2          = (float*)alloc((size_t)2 * NV * 128 * 4);
  float* per_graph   = (float*)alloc((size_t)NG * 8 * 4);        // round-zeroed region end
  long   zero_rgn_n  = ((long)2 * NV * 128) * 2 + NG * 8;
  float* lit_deg     = (float*)alloc((size_t)2 * NV * 4);
  float* dw          = (float*)alloc((size_t)2 * NV * 4);
  float* vdw         = (float*)alloc((size_t)NV * 4);
  // f16 activation buffers (contiguous; zeroed once so pad columns stay 0)
  _Float16* h_begin = (_Float16*)(ws + off);
  _Float16* v1h   = (_Float16*)alloc((size_t)NV * 160 * 2);
  _Float16* qh1   = (_Float16*)alloc((size_t)NV * 160 * 2);
  _Float16* varh  = (_Float16*)alloc((size_t)NV * 128 * 2);
  _Float16* lh1   = (_Float16*)alloc((size_t)NV * 512 * 2);
  _Float16* lh2   = (_Float16*)alloc((size_t)NV * 512 * 2);
  _Float16* cuh   = (_Float16*)alloc((size_t)NC * 384 * 2);
  _Float16* ch1   = (_Float16*)alloc((size_t)NC * 224 * 2);
  _Float16* unith = (_Float16*)alloc((size_t)NV * 512 * 2);
  _Float16* uh1   = (_Float16*)alloc((size_t)NV * 256 * 2);
  _Float16* uh2   = (_Float16*)alloc((size_t)NV * 256 * 2);
  _Float16* vh1   = (_Float16*)alloc((size_t)NV * 128 * 2);
  long h_act_n = ((_Float16*)(ws + off)) - h_begin;
  // f16 weights (padded [Kpad][Npad], fully written by convert)
  _Float16* Wq0 = (_Float16*)alloc((size_t)160 * 160 * 2);
  _Float16* Wq1 = (_Float16*)alloc((size_t)160 * 128 * 2);
  _Float16* Wl0 = (_Float16*)alloc((size_t)128 * 512 * 2);
  _Float16* Wl1 = (_Float16*)alloc((size_t)512 * 512 * 2);
  _Float16* Wl2 = (_Float16*)alloc((size_t)512 * 256 * 2);
  _Float16* Wc0 = (_Float16*)alloc((size_t)384 * 208 * 2);
  _Float16* Wc1 = (_Float16*)alloc((size_t)224 * 256 * 2);
  _Float16* Wu0 = (_Float16*)alloc((size_t)512 * 240 * 2);
  _Float16* Wu1 = (_Float16*)alloc((size_t)256 * 240 * 2);
  _Float16* Wu2 = (_Float16*)alloc((size_t)256 * 128 * 2);
  _Float16* Wv0 = (_Float16*)alloc((size_t)128 * 128 * 2);
  _Float16* Wv1 = (_Float16*)alloc((size_t)128 * 16 * 2);

  auto fill32 = [&](float* p, float v, long n) {
    fill_f32_kernel<<<dim3((unsigned)((n + 255) / 256)), dim3(256), 0, stream>>>(p, v, n);
  };
  auto convw = [&](const float* src, _Float16* dst, int K, int N, int Kp, int Np) {
    long tot = (long)Kp * Np;
    convert_w_kernel<<<dim3((unsigned)((tot + 255) / 256)), dim3(256), 0, stream>>>(src, dst, K, N, Kp, Np);
  };
  auto gemm = [&](const _Float16* A, int lda, const _Float16* W, int ldb, const float* bias,
                  float* C32, int ldc32, _Float16* C16, int ldc16, int M, int N, int K, int act) {
    dim3 grid((unsigned)((M + 63) / 64), (unsigned)((N + 127) / 128));
    gemm_f16w<<<grid, dim3(256), 0, stream>>>(A, lda, W, ldb, bias, C32, ldc32, C16, ldc16, M, N, K, act);
  };

  // ---- once-per-call init ----
  fill_f16_kernel<<<dim3((unsigned)((h_act_n + 255) / 256)), dim3(256), 0, stream>>>(h_begin, h_act_n);
  fill32(variables, 1.f, (long)NV * 128);
  fill32(clause_st, 1.f, (long)NC * 128);
  fill32(lit_deg, 0.f, 2 * NV);
  fill32(out + (long)NV * 8, 0.f, 1);  // loss accumulator slot
  deg_scatter_kernel<<<dim3((NE + 255) / 256), dim3(256), 0, stream>>>(lit_idx, lit_deg);
  deg_weights_kernel<<<dim3((2 * NV + 255) / 256), dim3(256), 0, stream>>>(lit_deg, dw, vdw);
  convw(qw0, Wq0, 132, 153, 160, 160);
  convw(qw1, Wq1, 153, 128, 160, 128);
  convw(lw0, Wl0, 128, 512, 128, 512);
  convw(lw1, Wl1, 512, 512, 512, 512);
  convw(lw2, Wl2, 512, 256, 512, 256);
  convw(cw0, Wc0, 384, 204, 384, 208);
  convw(cw1, Wc1, 204, 256, 224, 256);
  convw(uw0, Wu0, 512, 230, 512, 240);
  convw(uw1, Wu1, 230, 230, 256, 240);
  convw(uw2, Wu2, 230, 128, 256, 128);
  convw(ow0, Wv0, 128, 128, 128, 128);
  convw(ow1, Wv1, 128, 8, 128, 16);

  const dim3 B256(256);
  const dim3 Gv((NV * 128 + 255) / 256);
  const dim3 Gc((NC * 128 + 255) / 256);
  const dim3 Ge((unsigned)(((long)NE * 128 + 255) / 256));

  for (int t = 0; t < RNDS; ++t) {
    const float* noise_t = noise + (size_t)t * NV * 4;
    fill32(S, 0.f, zero_rgn_n);  // S, S2, per_graph contiguous
    build_v1_kernel<<<Gv, B256, 0, stream>>>(variables, noise_t, v1h, varh, unith);
    // variables_query
    gemm(v1h, 160, Wq0, 160, qb0, nullptr, 0, qh1, 160, NV, 153, 160, 1);
    gemm(qh1, 160, Wq1, 128, qb1, query, 128, nullptr, 0, NV, 128, 160, 0);
    // clause values + grad
    clause_val_kernel<<<Gc, B256, 0, stream>>>(lit_idx, query, clv, cuh);
    scatter_cl_kernel<<<Ge, B256, 0, stream>>>(lit_idx, clv, S);
    grad_kernel<<<Gv, B256, 0, stream>>>(query, S, vdw, unith);
    // lit_mlp
    gemm(varh, 128, Wl0, 512, lb0, nullptr, 0, lh1, 512, NV, 512, 128, 1);
    gemm(lh1, 512, Wl1, 512, lb1, nullptr, 0, lh2, 512, NV, 512, 512, 1);
    gemm(lh2, 512, Wl2, 256, lb2, var_msg, 256, nullptr, 0, NV, 256, 512, 0);
    // clause unit + clause_mlp
    build_cu_kernel<<<Gc, B256, 0, stream>>>(lit_idx, clause_st, var_msg, cuh);
    gemm(cuh, 384, Wc0, 208, cb0, nullptr, 0, ch1, 224, NC, 204, 384, 1);
    gemm(ch1, 224, Wc1, 256, cb1, clause_data, 256, nullptr, 0, NC, 256, 224, 0);
    // clause pair-norm + residual
    pair_norm_kernel<<<dim3(NG), dim3(128), 0, stream>>>(clause_data + 128, 256, clause_st, clause_st, CPG);
    // vloss
    scatter_vloss_kernel<<<Ge, B256, 0, stream>>>(lit_idx, clause_data, S2);
    vloss_kernel<<<Gv, B256, 0, stream>>>(S2, dw, unith);
    // update_gate
    gemm(unith, 512, Wu0, 240, ub0, nullptr, 0, uh1, 256, NV, 230, 512, 1);
    gemm(uh1, 256, Wu1, 240, ub1, nullptr, 0, uh2, 256, NV, 230, 256, 1);
    gemm(uh2, 256, Wu2, 128, ub2, newvars, 128, nullptr, 0, NV, 128, 256, 0);
    pair_norm_kernel<<<dim3(NG), dim3(128), 0, stream>>>(newvars, 128, variables, variables, VPG);
    // variables_output -> logits (write into d_out region)
    cvt_h_kernel<<<Gv, B256, 0, stream>>>(variables, varh, (long)NV * 128);
    gemm(varh, 128, Wv0, 128, ob0, nullptr, 0, vh1, 128, NV, 128, 128, 1);
    gemm(vh1, 128, Wv1, 16, ob1, out, 8, nullptr, 0, NV, 8, 128, 0);
    // loss
    loss_clause_kernel<<<dim3((NC + 255) / 256), B256, 0, stream>>>(lit_idx, out, per_graph);
    loss_fin_kernel<<<dim3(1), dim3(32), 0, stream>>>(per_graph, out + (long)NV * 8);
  }
}
